// EmbeddingMatcher_19129784336901
// MI455X (gfx1250) — matrix operand order repaired and checked by reading, not compiled, not run on hardware
//
#include <hip/hip_runtime.h>
#include <hip/hip_bf16.h>

// VQ nearest-codebook search for MI455X (gfx1250, wave32, WMMA).
// B=16, D=64, L=8192, K=512 ; N = B*L = 131072 tokens.

typedef __attribute__((ext_vector_type(16))) __bf16 v16bf;
typedef __attribute__((ext_vector_type(8)))  float  v8f;

#define VQ_B 16
#define VQ_D 64
#define VQ_L 8192
#define VQ_K 512
#define TOK_PER_BLOCK 128                         // 8 waves * 16 tokens
#define NBLOCKS ((VQ_B * VQ_L) / TOK_PER_BLOCK)   // 1024
#define BFRAG_ELEMS (32 * 2 * 32 * 16)            // 32 col-tiles * 2 k-chunks * 32 lanes * 16 bf16
#define BFRAG_BYTES (BFRAG_ELEMS * 2)             // 64 KB
#define XTILE_BYTES (TOK_PER_BLOCK * VQ_D * 2)    // 16 KB
#define ESQ_BYTES   (VQ_K * 4)                    // 2 KB
#define IDX_BYTES   (TOK_PER_BLOCK * 4)           // 512 B
#define SMEM_BYTES  (BFRAG_BYTES + XTILE_BYTES + ESQ_BYTES + IDX_BYTES)

// ---------------------------------------------------------------------------
// Prep: swizzle codebook (D x K fp32) into WMMA B-fragment order (bf16) and
// compute ||e_k||^2. B 32x16 bf16 order used here (contiguous k): lane -> col N=lane&15,
// half-wave g=lane>>4 gives K offset 16, element e -> K = g*16 + e.
// ---------------------------------------------------------------------------
__global__ void vq_prep(const float* __restrict__ emb,
                        __bf16* __restrict__ bfrag,
                        float* __restrict__ esq) {
    int tid    = blockIdx.x * blockDim.x + threadIdx.x;
    int stride = gridDim.x * blockDim.x;
    for (int i = tid; i < BFRAG_ELEMS; i += stride) {
        int e    = i & 15;
        int lane = (i >> 4) & 31;
        int f    = i >> 9;          // fragment = c*2 + kc
        int c    = f >> 1;
        int kc   = f & 1;
        int d    = kc * 32 + (lane >> 4) * 16 + e;   // gemm-K = feature dim
        int k    = c * 16 + (lane & 15);             // gemm-N = code index
        bfrag[i] = (__bf16)emb[d * VQ_K + k];
    }
    for (int k = tid; k < VQ_K; k += stride) {
        float s = 0.0f;
        for (int d = 0; d < VQ_D; ++d) {
            float v = emb[d * VQ_K + k];
            s += v * v;
        }
        esq[k] = s;
    }
}

// ---------------------------------------------------------------------------
// Main: per block, 128 tokens vs all 512 codes via v_wmma_f32_16x16x32_bf16.
// B fragments staged global->LDS once per block with async-to-LDS DMA
// (global_load_async_to_lds_b128, ASYNCcnt), then served from LDS to all
// 8 waves — cuts L2->WGP traffic 8x vs per-wave global reads.
// ---------------------------------------------------------------------------
__global__ __launch_bounds__(256)
void vq_main(const float* __restrict__ fv,
             const float* __restrict__ emb,
             const __bf16* __restrict__ bfrag,
             const float* __restrict__ esq,
             float* __restrict__ out_emb,
             float* __restrict__ out_idx) {
    extern __shared__ char smem[];
    __bf16* bfrag_s = (__bf16*)smem;                          // 64 KB
    __bf16* xtile   = (__bf16*)(smem + BFRAG_BYTES);          // 16 KB
    float*  esq_s   = (float*)(smem + BFRAG_BYTES + XTILE_BYTES);
    int*    idx_s   = (int*)(smem + BFRAG_BYTES + XTILE_BYTES + ESQ_BYTES);

    const int tid = threadIdx.x;
    const int n0  = blockIdx.x * TOK_PER_BLOCK;      // 128-token tile, within one batch
    const int b   = n0 >> 13;                        // / 8192
    const int l0  = n0 & (VQ_L - 1);
    const float* fvb = fv + (size_t)b * VQ_D * VQ_L;

    // --- Async DMA: B fragments global -> LDS (64 KB, 16B/lane/iter). ---
    {
        unsigned lds_base = (unsigned)(uintptr_t)bfrag_s;     // low 32b = LDS offset
        #pragma unroll
        for (int it = 0; it < 16; ++it) {
            unsigned byte_off = (unsigned)(it * 256 + tid) * 16u;
            unsigned lds_addr = lds_base + byte_off;
            asm volatile("global_load_async_to_lds_b128 %0, %1, %2"
                         :
                         : "v"(lds_addr), "v"(byte_off), "s"(bfrag)
                         : "memory");
        }
    }

    // Stage X tile (coalesced along L), convert fp32 -> bf16.
    #pragma unroll 4
    for (int it = 0; it < 32; ++it) {
        int lin = it * 256 + tid;
        int d = lin >> 7, t = lin & 127;
        xtile[t * VQ_D + d] = (__bf16)fvb[d * VQ_L + l0 + t];
    }
    for (int i = tid; i < VQ_K; i += 256) esq_s[i] = esq[i];

    asm volatile("s_wait_asynccnt 0" ::: "memory");
    __syncthreads();

    const int wave = tid >> 5;
    const int lane = tid & 31;
    const int g    = lane >> 4;      // half-wave
    const int m    = lane & 15;      // A row / C column within tile

    // Build A fragments (16-bit A 16x32, same contiguous k order as B):
    // element e -> K = 16*g + e
    v16bf a0, a1;
    const __bf16* xr = xtile + (wave * 16 + m) * VQ_D;
    #pragma unroll
    for (int e = 0; e < 16; ++e) {
        int kk = g * 16 + e;
        a0[e] = xr[kk];
        a1[e] = xr[32 + kk];
    }

    float best[8];
    int   bidx[8];
    #pragma unroll
    for (int r = 0; r < 8; ++r) { best[r] = 3.4e38f; bidx[r] = 0; }

    // 32 column tiles of 16 codes; 2 WMMAs each (K=64 as two K=32 chunks).
    #pragma unroll 4
    for (int c = 0; c < 32; ++c) {
        const v16bf b0 = *(const v16bf*)(bfrag_s + ((c * 2 + 0) * 32 + lane) * 16);
        const v16bf b1 = *(const v16bf*)(bfrag_s + ((c * 2 + 1) * 32 + lane) * 16);
        v8f acc = {};
        acc = __builtin_amdgcn_wmma_f32_16x16x32_bf16(false, a0, false, b0,
                                                      (short)0, acc, false, false);
        acc = __builtin_amdgcn_wmma_f32_16x16x32_bf16(false, a1, false, b1,
                                                      (short)0, acc, false, false);
        const int   kcol = c * 16 + m;
        const float eq   = esq_s[kcol];
        #pragma unroll
        for (int r = 0; r < 8; ++r) {
            float dist = eq - 2.0f * acc[r];   // ||x||^2 dropped: constant per row
            if (dist < best[r]) { best[r] = dist; bidx[r] = kcol; }
        }
    }

    // Min-reduce across the 16 lanes of each half-wave (row M = r + 8*g).
    // Tie -> lower index (jnp.argmin picks first).
    #pragma unroll
    for (int off = 1; off < 16; off <<= 1) {
        #pragma unroll
        for (int r = 0; r < 8; ++r) {
            float ov = __shfl_xor(best[r], off, 32);
            int   oi = __shfl_xor(bidx[r], off, 32);
            if (ov < best[r] || (ov == best[r] && oi < bidx[r])) {
                best[r] = ov; bidx[r] = oi;
            }
        }
    }
    if (m == 0) {
        #pragma unroll
        for (int r = 0; r < 8; ++r) idx_s[wave * 16 + g * 8 + r] = bidx[r];
    }
    __syncthreads();

    // Emit indices (as float: d_out is one fp32 buffer) ...
    for (int t = tid; t < TOK_PER_BLOCK; t += 256)
        out_idx[(size_t)b * VQ_L + l0 + t] = (float)idx_s[t];

    // ... and gather winning codebook rows, coalesced along L.
    #pragma unroll 4
    for (int it = 0; it < 32; ++it) {
        int lin = it * 256 + tid;
        int d = lin >> 7, t = lin & 127;
        out_emb[(size_t)b * VQ_D * VQ_L + (size_t)d * VQ_L + l0 + t] =
            emb[d * VQ_K + idx_s[t]];
    }
}

extern "C" void kernel_launch(void* const* d_in, const int* in_sizes, int n_in,
                              void* d_out, int out_size, void* d_ws, size_t ws_size,
                              hipStream_t stream) {
    const float* fv  = (const float*)d_in[0];   // (B, D, L) fp32
    const float* emb = (const float*)d_in[1];   // (D, K)    fp32

    __bf16* bfrag = (__bf16*)d_ws;                                   // 64 KB
    float*  esq   = (float*)((char*)d_ws + BFRAG_BYTES);             // +2 KB

    float* out_emb = (float*)d_out;                                  // (B, D, L)
    float* out_idx = out_emb + (size_t)VQ_B * VQ_D * VQ_L;           // (B, L)

    vq_prep<<<32, 256, 0, stream>>>(emb, bfrag, esq);
    vq_main<<<NBLOCKS, 256, SMEM_BYTES, stream>>>(fv, emb, bfrag, esq, out_emb, out_idx);
}
